// SCAM_9354438770840
// MI455X (gfx1250) — matrix operand order repaired and checked by reading, not compiled, not run on hardware
//
#include <hip/hip_runtime.h>

typedef __attribute__((ext_vector_type(16))) _Float16 v16h;
typedef __attribute__((ext_vector_type(8)))  float    v8f;

// Problem constants (match reference)
constexpr int Bn = 4, Cc = 64, Hh = 128, Ww = 512;
constexpr int HW  = Hh * Ww;        // 65536
constexpr int CHW = Cc * HW;        // 4194304
constexpr int QS  = 72;             // row stride (halves) for Q/K buffers (64 + 8 pad)
constexpr int VS  = Ww + 8;         // 520, row stride (halves) for transposed V
constexpr float EPS   = 1e-6f;
constexpr float SCALE = 0.125f;     // 64^-0.5

// ---- fragment loaders (LDS) -------------------------------------------------
// B-fragment f16 (K=32 chunk): lane holds col N=lane%16, halves h <-> K = 16*(lane/16)+h
__device__ __forceinline__ v16h ld_b16(const _Float16* p) {
  v16h r;
  ((uint4*)&r)[0] = ((const uint4*)p)[0];
  ((uint4*)&r)[1] = ((const uint4*)(p + 8))[0];
  return r;
}
// A-fragment f16 (K=32 chunk): lane holds row M=lane%16,
// halves h <-> K = kbase + 16*g + h
__device__ __forceinline__ v16h ld_afrag(const _Float16* row, int kbase, int g) {
  v16h r;
  ((uint4*)&r)[0] = ((const uint4*)(row + kbase + 16 * g))[0];
  ((uint4*)&r)[1] = ((const uint4*)(row + kbase + 16 * g + 8))[0];
  return r;
}
__device__ __forceinline__ v8f wmma16(v16h a, v16h b, v8f c) {
  return __builtin_amdgcn_wmma_f32_16x16x32_f16(false, a, false, b, (short)0, c,
                                                false, false);
}

__global__ __launch_bounds__(256, 1)
void scam_fused_kernel(const float* __restrict__ x_l, const float* __restrict__ x_r,
                       const float* __restrict__ nlw, const float* __restrict__ nlb,
                       const float* __restrict__ nrw, const float* __restrict__ nrb,
                       const float* __restrict__ l1w, const float* __restrict__ l1b,
                       const float* __restrict__ r1w, const float* __restrict__ r1b,
                       const float* __restrict__ l2w, const float* __restrict__ l2b,
                       const float* __restrict__ r2w, const float* __restrict__ r2b,
                       const float* __restrict__ beta, const float* __restrict__ gamma,
                       float* __restrict__ out)
{
  // ---- LDS (~237 KB; CDNA5 allows up to 320 KB per workgroup) ----
  __shared__ __align__(16) _Float16 Qbuf[Ww * QS];       // 73728 B (staging + Q, in place)
  __shared__ __align__(16) _Float16 Kbuf[Ww * QS];       // 73728 B (staging + K, in place)
  __shared__ __align__(16) _Float16 Vt[Cc * VS];         // 66560 B (V stored transposed [c][w])
  __shared__ __align__(16) _Float16 Wbuf[3 * Cc * QS];   // 27648 B (Wq, Wk, Wv as f16)
  __shared__ float nrm[5 * Cc];                          // norm w/b (q,k) + out coef

  const int tid  = threadIdx.x;
  const int wave = tid >> 5;
  const int lane = tid & 31;
  const int lm   = lane & 15;
  const int g    = lane >> 4;

  const int bh  = blockIdx.x;
  const int b   = bh / Hh;
  const int h   = bh % Hh;
  const int dir = blockIdx.y;   // 0: r2l (out_l), 1: l2r (out_r)

  // role selection (k-source doubles as v-source)
  const float* xq  = dir ? x_r : x_l;
  const float* xk  = dir ? x_l : x_r;
  const float* Wq  = dir ? r1w : l1w;  const float* bq = dir ? r1b : l1b;
  const float* Wk  = dir ? l1w : r1w;  const float* bk = dir ? l1b : r1b;
  const float* Wv  = dir ? l2w : r2w;  const float* bv = dir ? l2b : r2b;
  const float* nqw = dir ? nrw : nlw;  const float* nqb = dir ? nrb : nlb;
  const float* nkw = dir ? nlw : nrw;  const float* nkb = dir ? nlb : nrb;
  const float* cf  = dir ? gamma : beta;
  float* outp = out + (dir ? (size_t)Bn * CHW : (size_t)0);

  const size_t base = (size_t)b * CHW + (size_t)h * Ww;

  // ---- Phase 0: stage weights (f16) + norm params, stage raw x_v -> Qbuf ----
  for (int i = tid; i < Cc * Cc; i += 256) {
    int o = i >> 6, c = i & 63;
    Wbuf[0 * Cc * QS + o * QS + c] = (_Float16)Wq[i];
    Wbuf[1 * Cc * QS + o * QS + c] = (_Float16)Wk[i];
    Wbuf[2 * Cc * QS + o * QS + c] = (_Float16)Wv[i];
  }
  if (tid < Cc) {
    nrm[tid]            = nqw[tid];
    nrm[Cc + tid]       = nqb[tid];
    nrm[2 * Cc + tid]   = nkw[tid];
    nrm[3 * Cc + tid]   = nkb[tid];
    nrm[4 * Cc + tid]   = cf[tid];
  }
  for (int w = tid; w < Ww; w += 256) {
    const float* src = xk + base + w;
    _Float16* dst = &Qbuf[w * QS];
    #pragma unroll
    for (int c = 0; c < Cc; ++c) dst[c] = (_Float16)src[(size_t)c * HW];
  }
  __syncthreads();

  // ---- Phase 1: V GEMM  (V = Xv * Wv^T + bv), stored transposed into Vt[c][w]
  for (int mt = 0; mt < 4; ++mt) {
    const int mrow = wave * 64 + mt * 16;
    v16h a0 = ld_afrag(&Qbuf[(mrow + lm) * QS], 0, g);
    v16h a1 = ld_afrag(&Qbuf[(mrow + lm) * QS], 32, g);
    #pragma unroll
    for (int nt = 0; nt < 4; ++nt) {
      v16h b0 = ld_b16(&Wbuf[2 * Cc * QS + (nt * 16 + lm) * QS + 16 * g]);
      v16h b1 = ld_b16(&Wbuf[2 * Cc * QS + (nt * 16 + lm) * QS + 32 + 16 * g]);
      v8f acc = {};
      acc = wmma16(a0, b0, acc);
      acc = wmma16(a1, b1, acc);
      const int   cout = nt * 16 + lm;        // D-frag: lane = cout, vgpr = w
      const float bb   = bv[cout];
      _Float16* vd = &Vt[cout * VS + mrow + 8 * g];
      #pragma unroll
      for (int v = 0; v < 8; ++v) vd[v] = (_Float16)(acc[v] + bb);
    }
  }
  __syncthreads();

  // ---- Phase 2: LayerNorm(x_q) -> Qbuf, LayerNorm(x_k) -> Kbuf (f16) ----
  for (int w = tid; w < Ww; w += 256) {
    {
      const float* s = xq + base + w;
      float sm = 0.f, ss = 0.f;
      #pragma unroll
      for (int c = 0; c < Cc; ++c) { float v = s[(size_t)c * HW]; sm += v; ss += v * v; }
      const float mu = sm * (1.f / Cc);
      const float rs = rsqrtf(ss * (1.f / Cc) - mu * mu + EPS);
      _Float16* d = &Qbuf[w * QS];
      #pragma unroll
      for (int c = 0; c < Cc; ++c)
        d[c] = (_Float16)((s[(size_t)c * HW] - mu) * rs * nrm[c] + nrm[Cc + c]);
    }
    {
      const float* s = xk + base + w;
      float sm = 0.f, ss = 0.f;
      #pragma unroll
      for (int c = 0; c < Cc; ++c) { float v = s[(size_t)c * HW]; sm += v; ss += v * v; }
      const float mu = sm * (1.f / Cc);
      const float rs = rsqrtf(ss * (1.f / Cc) - mu * mu + EPS);
      _Float16* d = &Kbuf[w * QS];
      #pragma unroll
      for (int c = 0; c < Cc; ++c)
        d[c] = (_Float16)((s[(size_t)c * HW] - mu) * rs * nrm[2 * Cc + c] + nrm[3 * Cc + c]);
    }
  }
  __syncthreads();

  // ---- Phase 3: Q and K GEMMs, in place (each wave owns 64 exclusive rows) ----
  for (int which = 0; which < 2; ++which) {
    _Float16* Bf = which ? Kbuf : Qbuf;
    const _Float16* Wb = &Wbuf[which * Cc * QS];
    const float* bias = which ? bk : bq;
    for (int mt = 0; mt < 4; ++mt) {
      const int mrow = wave * 64 + mt * 16;
      v16h a0 = ld_afrag(&Bf[(mrow + lm) * QS], 0, g);
      v16h a1 = ld_afrag(&Bf[(mrow + lm) * QS], 32, g);
      v8f accs[4];
      #pragma unroll
      for (int nt = 0; nt < 4; ++nt) {
        v16h b0 = ld_b16(&Wb[(nt * 16 + lm) * QS + 16 * g]);
        v16h b1 = ld_b16(&Wb[(nt * 16 + lm) * QS + 32 + 16 * g]);
        v8f acc = {};
        acc = wmma16(a0, b0, acc);
        acc = wmma16(a1, b1, acc);
        accs[nt] = acc;
      }
      #pragma unroll
      for (int nt = 0; nt < 4; ++nt) {
        const int   cout = nt * 16 + lm;
        const float bb   = bias[cout];
        #pragma unroll
        for (int v = 0; v < 8; ++v)
          Bf[(mrow + 8 * g + v) * QS + cout] = (_Float16)(accs[nt][v] + bb);
      }
    }
  }
  __syncthreads();

  // ---- Phase 4: attention, transposed form.  S^T = K*Q^T so lane = query. ----
  for (int qt = 0; qt < 4; ++qt) {
    const int qb = wave * 64 + qt * 16;
    // Q^T B-fragments (col = query)
    v16h qf0 = ld_b16(&Qbuf[(qb + lm) * QS + 16 * g]);
    v16h qf1 = ld_b16(&Qbuf[(qb + lm) * QS + 32 + 16 * g]);

    float m = -1e30f, l = 0.f;
    v8f zero = {};
    v8f oa[4] = {zero, zero, zero, zero};   // O^T accumulators: lane=query, vgpr=c

    for (int kb = 0; kb < 16; ++kb) {       // 32 keys per step
      const int kbase = kb * 32;
      v8f s0 = zero, s1 = zero;             // S^T tiles: keys kbase+0..15 / +16..31
      {
        v16h ka;
        ka = ld_afrag(&Kbuf[(kbase + lm) * QS], 0, g);       s0 = wmma16(ka, qf0, s0);
        ka = ld_afrag(&Kbuf[(kbase + lm) * QS], 32, g);      s0 = wmma16(ka, qf1, s0);
        ka = ld_afrag(&Kbuf[(kbase + 16 + lm) * QS], 0, g);  s1 = wmma16(ka, qf0, s1);
        ka = ld_afrag(&Kbuf[(kbase + 16 + lm) * QS], 32, g); s1 = wmma16(ka, qf1, s1);
      }
      float tm = -1e30f;
      #pragma unroll
      for (int i = 0; i < 8; ++i) {
        s0[i] *= SCALE; s1[i] *= SCALE;
        tm = fmaxf(tm, fmaxf(s0[i], s1[i]));
      }
      tm = fmaxf(tm, __shfl_xor(tm, 16, 32));   // other half of this query's keys
      const float mn   = fmaxf(m, tm);
      const float corr = __expf(m - mn);
      float rsum = 0.f;
      #pragma unroll
      for (int i = 0; i < 8; ++i) {
        s0[i] = __expf(s0[i] - mn);
        s1[i] = __expf(s1[i] - mn);
        rsum += s0[i] + s1[i];
      }
      rsum += __shfl_xor(rsum, 16, 32);
      l = l * corr + rsum;
      m = mn;
      #pragma unroll
      for (int ct = 0; ct < 4; ++ct)
        #pragma unroll
        for (int i = 0; i < 8; ++i) oa[ct][i] *= corr;       // lane-uniform rescale

      // Build P^T B-fragment: B halves need 16 consecutive keys per half-wave,
      // C-frags hold 8 -> swap halves across the two lane groups.
      v16h pb;
      #pragma unroll
      for (int i = 0; i < 8; ++i) {
        float o0 = __shfl_xor(s0[i], 16, 32);
        float o1 = __shfl_xor(s1[i], 16, 32);
        pb[i]     = (_Float16)(g ? o1    : s0[i]);   // keys kbase + 16g + i
        pb[8 + i] = (_Float16)(g ? s1[i] : o0);      // keys kbase + 16g + 8 + i
      }
      // O^T += V^T * P^T  (A from transposed V: contiguous along keys)
      #pragma unroll
      for (int ct = 0; ct < 4; ++ct) {
        v16h va = ld_afrag(&Vt[(ct * 16 + lm) * VS], kbase, g);
        oa[ct] = wmma16(va, pb, oa[ct]);
      }
    }

    // epilogue: normalize, *coef[c], + residual, write NCHW (coalesced over w)
    const float inv = 1.f / l;
    const int wq = qb + lm;
    #pragma unroll
    for (int ct = 0; ct < 4; ++ct)
      #pragma unroll
      for (int v = 0; v < 8; ++v) {
        const int c = ct * 16 + 8 * g + v;
        const size_t gi = base + (size_t)c * HW + wq;
        outp[gi] = xq[gi] + oa[ct][v] * inv * nrm[4 * Cc + c];
      }
  }
}

extern "C" void kernel_launch(void* const* d_in, const int* in_sizes, int n_in,
                              void* d_out, int out_size, void* d_ws, size_t ws_size,
                              hipStream_t stream) {
  (void)in_sizes; (void)n_in; (void)out_size; (void)d_ws; (void)ws_size;
  const float* x_l  = (const float*)d_in[0];
  const float* x_r  = (const float*)d_in[1];
  const float* nlw  = (const float*)d_in[2];
  const float* nlb  = (const float*)d_in[3];
  const float* nrw  = (const float*)d_in[4];
  const float* nrb  = (const float*)d_in[5];
  const float* l1w  = (const float*)d_in[6];
  const float* l1b  = (const float*)d_in[7];
  const float* r1w  = (const float*)d_in[8];
  const float* r1b  = (const float*)d_in[9];
  const float* l2w  = (const float*)d_in[10];
  const float* l2b  = (const float*)d_in[11];
  const float* r2w  = (const float*)d_in[12];
  const float* r2b  = (const float*)d_in[13];
  const float* beta = (const float*)d_in[14];
  const float* gmma = (const float*)d_in[15];
  float* out = (float*)d_out;

  dim3 grid(Bn * Hh, 2);   // 512 row-pairs x 2 directions
  dim3 block(256);         // 8 wave32 per workgroup
  scam_fused_kernel<<<grid, block, 0, stream>>>(x_l, x_r, nlw, nlb, nrw, nrb,
                                                l1w, l1b, r1w, r1b, l2w, l2b,
                                                r2w, r2b, beta, gmma, out);
}
